// SimplifiedGCN_29403346108559
// MI455X (gfx1250) — compile-verified, matrix operand order checked by reading, not run on hardware
//
#include <hip/hip_runtime.h>

typedef __attribute__((ext_vector_type(2)))  float        v2f;
typedef __attribute__((ext_vector_type(8)))  float        v8f;
typedef __attribute__((ext_vector_type(16))) _Float16     v16h;
typedef __attribute__((ext_vector_type(4)))  unsigned int v4u;
typedef __attribute__((ext_vector_type(4)))  int          v4i;
typedef __attribute__((ext_vector_type(8)))  int          v8i;

#define FEAT   512
#define HID    16
#define NCLS   10
#define ROWS_PER_BLOCK 64   // 4 waves x 16 rows

#if __has_builtin(__builtin_amdgcn_wmma_f32_16x16x4_f32)
#define USE_F32_WMMA 1
#else
#define USE_F32_WMMA 0
#endif

#if __has_builtin(__builtin_amdgcn_tensor_load_to_lds) && __has_builtin(__builtin_amdgcn_s_wait_tensorcnt)
#define USE_TDM 1
#else
#define USE_TDM 0
#endif

// ---------------- utility kernels ----------------

__global__ void fill_f32(float* __restrict__ p, float v, long long n) {
    long long i = (long long)blockIdx.x * blockDim.x + threadIdx.x;
    if (i < n) p[i] = v;
}

__global__ void deg_count(const int* __restrict__ dst, float* __restrict__ deg, long long E) {
    long long e = (long long)blockIdx.x * blockDim.x + threadIdx.x;
    if (e < E) atomicAdd(&deg[dst[e]], 1.0f);
}

__global__ void deg_to_dinv(float* __restrict__ deg, int n) {
    int i = blockIdx.x * blockDim.x + threadIdx.x;
    if (i < n) {
        float d = deg[i];
        deg[i] = (d > 0.0f) ? rsqrtf(d) : 0.0f;
    }
}

// ---------------- GEMM1: h1 = x @ W1 via WMMA, W1 staged in LDS via TDM ----------------
// Block = 128 threads = 4 waves; each wave computes a 16x16 tile (16 rows, all 16 hidden).
// W1 (512x16 f32 = 32 KB) is DMA'd into LDS once per block by wave 0's Tensor Data Mover op.
__global__ __launch_bounds__(128) void gemm1_wmma(const float* __restrict__ x,
                                                  const float* __restrict__ W,
                                                  float* __restrict__ h1, int N) {
    __shared__ float ldsW[FEAT * HID];   // 32 KB, LDS offset 0

    const int tid  = threadIdx.x;
    const int lane = tid & 31;
    const int wave = tid >> 5;

#if USE_TDM
    if (wave == 0) {
        // Tensor DMA descriptor: 2D tensor, row length 16 f32, 512 rows; tile = whole tensor.
        unsigned long long gaddr = (unsigned long long)(const void*)W;
        unsigned           laddr = (unsigned)(unsigned long long)(const void*)&ldsW[0]; // addr[31:0] == LDS offset
        v4u g0;
        g0[0] = 1u;                                            // count=1, user descriptor
        g0[1] = laddr;                                         // lds_addr (bytes)
        g0[2] = (unsigned)(gaddr & 0xffffffffu);               // global_addr[31:0]
        g0[3] = (unsigned)((gaddr >> 32) & 0x01ffffffu)        // global_addr[56:32]
              | 0x80000000u;                                   // type=2 ("image")
        v8i g1;
        g1[0] = (2 << 16);       // workgroup_mask=0, data_size=2 (4 bytes)
        g1[1] = (HID << 16);     // tensor_dim0 = 16        (bits 63:48)
        g1[2] = (FEAT << 16);    // tensor_dim1 = 512       (bits 95:80)
        g1[3] = (HID << 16);     // tile_dim0 = 16          (bits 127:112)
        g1[4] = FEAT;            // tile_dim1 = 512, tile_dim2 = 0
        g1[5] = HID;             // tensor_dim0_stride = 16 (low 32)
        g1[6] = 0;               // stride0 high, tensor_dim1_stride low
        g1[7] = 0;
        v4i z4 = {0, 0, 0, 0};                // groups 2/3 unused (<=2D tensor)
        v8i z8 = {0, 0, 0, 0, 0, 0, 0, 0};    // extra descriptor words (6-arg builtin form)
        __builtin_amdgcn_tensor_load_to_lds(g0, g1, z4, z4, z8, 0);
        __builtin_amdgcn_s_wait_tensorcnt(0);
    }
    __syncthreads();
#else
    // Fallback staging: cooperative copy.
    for (int i = tid; i < FEAT * HID; i += 128) ldsW[i] = W[i];
    __syncthreads();
#endif

    const int row0 = blockIdx.x * ROWS_PER_BLOCK + wave * 16;
    if (row0 >= N) return;

    const int m    = lane & 15;          // row (A) / col (B) within tile
    const int half = lane >> 4;          // 0: lanes 0-15, 1: lanes 16-31

    int row  = row0 + m;
    int rowc = row < N ? row : N - 1;    // clamp reads for tail tiles
    const float* xrow = x + (long long)rowc * FEAT;

    v8f acc0 = {};
    v8f acc1 = {};

#if USE_F32_WMMA
    // A 16x4 f32 layout: lane<16 holds K={0,1}, lane>=16 holds K={2,3}. B mirrors with N=lane&15.
    const int kk = half * 2;
    for (int k = 0; k < FEAT; k += 8) {
        v2f a0, b0, a1, b1;
        a0.x = xrow[k + kk];
        a0.y = xrow[k + kk + 1];
        a1.x = xrow[k + 4 + kk];
        a1.y = xrow[k + 4 + kk + 1];
        b0.x = ldsW[(k + kk) * HID + m];
        b0.y = ldsW[(k + kk + 1) * HID + m];
        b1.x = ldsW[(k + 4 + kk) * HID + m];
        b1.y = ldsW[(k + 4 + kk + 1) * HID + m];
        acc0 = __builtin_amdgcn_wmma_f32_16x16x4_f32(false, a0, false, b0,
                                                     (short)0, acc0, false, false);
        acc1 = __builtin_amdgcn_wmma_f32_16x16x4_f32(false, a1, false, b1,
                                                     (short)0, acc1, false, false);
    }
#else
    // Fallback: f16 inputs, f32 accumulate (16x16x32).
    const int kb = half * 8;
    for (int k = 0; k < FEAT; k += 32) {
        v16h a, b;
#pragma unroll
        for (int j = 0; j < 8; ++j) {
            a[j]     = (_Float16)xrow[k + kb + j];
            a[j + 8] = (_Float16)xrow[k + 16 + kb + j];
            b[j]     = (_Float16)ldsW[(k + kb + j) * HID + m];
            b[j + 8] = (_Float16)ldsW[(k + 16 + kb + j) * HID + m];
        }
        acc0 = __builtin_amdgcn_wmma_f32_16x16x32_f16(false, a, false, b,
                                                      (short)0, acc0, false, false);
    }
#endif

    // C/D layout: VGPR r -> M = r + 8*half, N = lane&15
    const int mbase = row0 + half * 8;
    if (row0 + 16 <= N) {
#pragma unroll
        for (int r = 0; r < 8; ++r)
            h1[(long long)(mbase + r) * HID + m] = acc0[r] + acc1[r];
    } else {
#pragma unroll
        for (int r = 0; r < 8; ++r) {
            int mr = mbase + r;
            if (mr < N) h1[(long long)mr * HID + m] = acc0[r] + acc1[r];
        }
    }
}

// ---------------- edge aggregation (gather * norm -> scatter atomicAdd) ----------------
template <int F>
__global__ void aggregate_k(const float* __restrict__ h,
                            const int* __restrict__ src, const int* __restrict__ dst,
                            const float* __restrict__ dinv,
                            float* __restrict__ out, long long E) {
    long long t = (long long)blockIdx.x * blockDim.x + threadIdx.x;
    long long e = t / F;
    if (e >= E) return;
    int f = (int)(t - e * F);
    int s = src[e];
    int d = dst[e];
    float norm = dinv[s] * dinv[d];
    atomicAdd(&out[(long long)d * F + f], h[(long long)s * F + f] * norm);
}

// ---------------- layer-1 finalize: self-loop + bias + ReLU (in place in agg1) ----------------
__global__ void finalize1_k(float* __restrict__ agg1, const float* __restrict__ h1,
                            const float* __restrict__ dinv, const float* __restrict__ b1,
                            int N) {
    int i = blockIdx.x * blockDim.x + threadIdx.x;
    if (i >= N * HID) return;
    int node = i >> 4;
    int f = i & (HID - 1);
    float d = dinv[node];
    float v = agg1[i] + h1[i] * d * d + b1[f];
    agg1[i] = v > 0.0f ? v : 0.0f;
}

// ---------------- GEMM2 (K=16, tiny): h2 = h @ W2 ----------------
__global__ void gemm2_k(const float* __restrict__ h, const float* __restrict__ W2,
                        float* __restrict__ h2, int N) {
    int i = blockIdx.x * blockDim.x + threadIdx.x;
    if (i >= N * NCLS) return;
    int node = i / NCLS;
    int c = i - node * NCLS;
    const float* hr = h + (long long)node * HID;
    float s = 0.0f;
#pragma unroll
    for (int k = 0; k < HID; ++k) s = fmaf(hr[k], W2[k * NCLS + c], s);
    h2[i] = s;
}

// ---------------- layer-2 finalize: self-loop + bias into d_out ----------------
__global__ void finalize2_k(float* __restrict__ out, const float* __restrict__ h2,
                            const float* __restrict__ dinv, const float* __restrict__ b2,
                            int N) {
    int i = blockIdx.x * blockDim.x + threadIdx.x;
    if (i >= N * NCLS) return;
    int node = i / NCLS;
    int c = i - node * NCLS;
    float d = dinv[node];
    out[i] += h2[i] * d * d + b2[c];
}

// ---------------- launch ----------------
extern "C" void kernel_launch(void* const* d_in, const int* in_sizes, int n_in,
                              void* d_out, int out_size, void* d_ws, size_t ws_size,
                              hipStream_t stream) {
    const float* x  = (const float*)d_in[0];
    const int*   ei = (const int*)d_in[1];
    const float* W1 = (const float*)d_in[2];
    const float* b1 = (const float*)d_in[3];
    const float* W2 = (const float*)d_in[4];
    const float* b2 = (const float*)d_in[5];
    float* out = (float*)d_out;

    const int       N = in_sizes[0] / FEAT;       // 100000
    const long long E = in_sizes[1] / 2;          // 3200000
    const int* srcI = ei;
    const int* dstI = ei + E;

    float* ws   = (float*)d_ws;
    float* dinv = ws;                                   // N
    float* h1   = dinv + N;                             // N*HID
    float* agg1 = h1 + (long long)N * HID;              // N*HID  (becomes h after finalize1)
    float* h2   = agg1 + (long long)N * HID;            // N*NCLS

    const int B = 256;
    auto blk = [](long long n, int b) { return (unsigned)((n + b - 1) / b); };

    // 1) init: deg=1 (self loop), agg1=0, out=0
    fill_f32<<<blk(N, B), B, 0, stream>>>(dinv, 1.0f, N);
    fill_f32<<<blk((long long)N * HID, B), B, 0, stream>>>(agg1, 0.0f, (long long)N * HID);
    fill_f32<<<blk((long long)N * NCLS, B), B, 0, stream>>>(out, 0.0f, (long long)N * NCLS);

    // 2) degrees -> dinv
    deg_count<<<blk(E, B), B, 0, stream>>>(dstI, dinv, E);
    deg_to_dinv<<<blk(N, B), B, 0, stream>>>(dinv, N);

    // 3) GEMM1 via WMMA (4 waves/block, W1 via TDM into LDS)
    gemm1_wmma<<<(unsigned)((N + ROWS_PER_BLOCK - 1) / ROWS_PER_BLOCK), 128, 0, stream>>>(x, W1, h1, N);

    // 4) layer-1 edge aggregation
    aggregate_k<HID><<<blk(E * HID, B), B, 0, stream>>>(h1, srcI, dstI, dinv, agg1, E);

    // 5) finalize layer 1 (self loop + bias + ReLU) -> agg1 holds h
    finalize1_k<<<blk((long long)N * HID, B), B, 0, stream>>>(agg1, h1, dinv, b1, N);

    // 6) GEMM2
    gemm2_k<<<blk((long long)N * NCLS, B), B, 0, stream>>>(agg1, W2, h2, N);

    // 7) layer-2 edge aggregation into out
    aggregate_k<NCLS><<<blk(E * NCLS, B), B, 0, stream>>>(h2, srcI, dstI, dinv, out, E);

    // 8) finalize layer 2 (self loop + bias)
    finalize2_k<<<blk((long long)N * NCLS, B), B, 0, stream>>>(out, h2, dinv, b2, N);
}
